// Siglip2Attention_43473658970146
// MI455X (gfx1250) — matrix operand order repaired and checked by reading, not compiled, not run on hardware
//
#include <hip/hip_runtime.h>

#define EMBED 1152
#define NHEAD 16
#define HDIM 72
#define HPAD 96
#define TTOK 4096
#define NSEG 8

typedef __attribute__((ext_vector_type(16))) __bf16 v16bf;
typedef __attribute__((ext_vector_type(8)))  float  v8f;

__device__ __forceinline__ v8f wmma_bf16(v16bf a, v16bf b, v8f c) {
  return __builtin_amdgcn_wmma_f32_16x16x32_bf16(false, a, false, b, (short)0, c, false, false);
}

// Assemble the 16x32 bf16 A-operand fragment for lane (m=row, hi) from row-major global:
// elements 0..15 = X[row][kc + hi*16 + 0..15]
__device__ __forceinline__ v16bf load_A(const __bf16* __restrict__ X, int ldx,
                                        int row, int kc, int hi) {
  const __bf16* base = X + (size_t)row * ldx + kc + hi * 16;
  union { v16bf v; float4 f[2]; } u;
  u.f[0] = *(const float4*)(base);
  u.f[1] = *(const float4*)(base + 8);
  return u.v;
}

// ---------------- fp32 -> bf16 convert ----------------
__global__ void f32_to_bf16_kernel(const float* __restrict__ src,
                                   __bf16* __restrict__ dst, size_t n) {
  size_t i = (size_t)blockIdx.x * blockDim.x + threadIdx.x;
  size_t stride = (size_t)gridDim.x * blockDim.x;
  for (; i < n; i += stride) dst[i] = (__bf16)src[i];
}

// ---------------- LDS-tiled WMMA GEMM: C[M,N] = X[M,K] * W[N,K]^T + bias ----------------
// block = 256 threads (8 waves) computing a 128x128 tile; wave tile 32(M) x 64(N).
// Per K-step, the block stages A(128x32) and B(128x32) bf16 panels in LDS.
#define LDT 40  // padded LDS row stride (halves): 80B/row -> conflict-free b128 reads

template <bool OUT_BF16>
__global__ __launch_bounds__(256, 1)
void gemm_lds_kernel(const __bf16* __restrict__ X,
                     const __bf16* __restrict__ W,
                     const float* __restrict__ bias,
                     void* __restrict__ Cout,
                     int M, int N, int K) {
  __shared__ __bf16 sA[128 * LDT];
  __shared__ __bf16 sB[128 * LDT];

  const int tid  = threadIdx.x;
  const int wave = tid >> 5;
  const int lane = tid & 31;
  const int nlo  = lane & 15;
  const int hi   = lane >> 4;
  const int wm   = wave >> 1;   // 0..3  -> M sub-tile
  const int wn   = wave & 1;    // 0..1  -> N sub-tile
  const int mt0  = blockIdx.y * 128;
  const int nt0  = blockIdx.x * 128;

  v8f acc[2][4];
#pragma unroll
  for (int i = 0; i < 2; ++i)
#pragma unroll
    for (int j = 0; j < 4; ++j) acc[i][j] = (v8f){0.f,0.f,0.f,0.f,0.f,0.f,0.f,0.f};

  for (int kc = 0; kc < K; kc += 32) {
    __syncthreads();  // previous iteration's LDS reads complete before overwrite
    // cooperative staging: 512 16B-chunks per panel, 2 per thread per panel
#pragma unroll
    for (int half = 0; half < 2; ++half) {
      const int id  = tid + half * 256;
      const int row = id >> 2;
      const int cs  = (id & 3) * 8;
      const float4 av = *(const float4*)(X + (size_t)(mt0 + row) * K + kc + cs);
      *(float4*)(sA + row * LDT + cs) = av;
      const float4 bv = *(const float4*)(W + (size_t)(nt0 + row) * K + kc + cs);
      *(float4*)(sB + row * LDT + cs) = bv;
      if (kc + 32 < K) {
        __builtin_prefetch(X + (size_t)(mt0 + row) * K + kc + 32 + cs, 0, 0);
        __builtin_prefetch(W + (size_t)(nt0 + row) * K + kc + 32 + cs, 0, 0);
      }
    }
    __syncthreads();

    // A fragments from LDS
    union { v16bf v; float4 f[2]; } ua0, ua1;
    {
      const __bf16* pa0 = sA + (wm * 32 + nlo) * LDT;
      const __bf16* pa1 = sA + (wm * 32 + 16 + nlo) * LDT;
      ua0.f[0] = *(const float4*)(pa0 + hi * 8);
      ua0.f[1] = *(const float4*)(pa0 + 16 + hi * 8);
      ua1.f[0] = *(const float4*)(pa1 + hi * 8);
      ua1.f[1] = *(const float4*)(pa1 + 16 + hi * 8);
    }
#pragma unroll
    for (int j = 0; j < 4; ++j) {
      union { v16bf v; float4 f[2]; } ub;
      const __bf16* pb = sB + (wn * 64 + j * 16 + nlo) * LDT + hi * 8;
      ub.f[0] = *(const float4*)(pb);
      ub.f[1] = *(const float4*)(pb + 16);
      acc[0][j] = wmma_bf16(ua0.v, ub.v, acc[0][j]);
      acc[1][j] = wmma_bf16(ua1.v, ub.v, acc[1][j]);
    }
  }

#pragma unroll
  for (int i = 0; i < 2; ++i) {
#pragma unroll
    for (int j = 0; j < 4; ++j) {
      const int n = nt0 + wn * 64 + j * 16 + nlo;
      const float bv = bias ? bias[n] : 0.f;
#pragma unroll
      for (int r = 0; r < 8; ++r) {
        const int m = mt0 + wm * 32 + i * 16 + hi * 8 + r;
        const float v = acc[i][j][r] + bv;
        if (OUT_BF16)
          ((__bf16*)Cout)[(size_t)m * N + n] = (__bf16)v;
        else
          ((float*)Cout)[(size_t)m * N + n] = v;
      }
    }
  }
}

// ---------------- pack: qkv bf16 [T][3456] -> Qb,Kb [H][T][96], Vt [H][96][T] ----------------
__global__ void pack_qkv_kernel(const __bf16* __restrict__ qkv,
                                __bf16* __restrict__ Qb,
                                __bf16* __restrict__ Kb,
                                __bf16* __restrict__ Vt,
                                float scale) {
  const int d = threadIdx.x;   // 0..95
  const int t = blockIdx.x;    // 0..4095
  const int h = blockIdx.y;    // 0..15
  __bf16 qv = (__bf16)0.f, kv = (__bf16)0.f, vv = (__bf16)0.f;
  if (d < HDIM) {
    const __bf16* row = qkv + (size_t)t * (3 * EMBED);
    qv = (__bf16)((float)row[h * HDIM + d] * scale);
    kv = row[EMBED + h * HDIM + d];
    vv = row[2 * EMBED + h * HDIM + d];
  }
  const size_t qi = ((size_t)h * TTOK + t) * HPAD + d;
  Qb[qi] = qv;
  Kb[qi] = kv;
  Vt[((size_t)h * HPAD + d) * TTOK + t] = vv;
}

// ---------------- flash attention: one wave per (16-query tile, head) ----------------
__global__ __launch_bounds__(32, 1)
void attn_kernel(const __bf16* __restrict__ Qb,
                 const __bf16* __restrict__ Kb,
                 const __bf16* __restrict__ Vt,
                 const int* __restrict__ cu,
                 __bf16* __restrict__ attnb) {
  const int qt = blockIdx.x;   // 0..255
  const int h  = blockIdx.y;   // 0..15
  const int lane = threadIdx.x & 31;
  const int nlo = lane & 15;
  const int hi  = lane >> 4;

  __shared__ __bf16 sP[16][32];

  int cuv[NSEG + 1];
#pragma unroll
  for (int i = 0; i <= NSEG; ++i) cuv[i] = cu[i];

  auto segof = [&](int t) {
    int s = 0;
#pragma unroll
    for (int i = 1; i < NSEG; ++i) s += (t >= cuv[i]);
    return s;
  };

  const int q0 = qt * 16;
  const int seg_first = segof(q0);
  const int seg_last  = segof(q0 + 15);
  const int sfs    = cuv[seg_first];      // first valid key
  const int bnd    = cuv[seg_last];       // boundary between the (<=2) segments
  const int kend   = cuv[seg_last + 1];   // one past last valid key
  const int kstart = sfs & ~31;

  bool rq[8];
#pragma unroll
  for (int r = 0; r < 8; ++r) rq[r] = (q0 + hi * 8 + r) >= bnd;

  const __bf16* Qh = Qb + ((size_t)h * TTOK + q0) * HPAD;
  v16bf aq[3];
#pragma unroll
  for (int c = 0; c < 3; ++c) aq[c] = load_A(Qh, HPAD, nlo, c * 32, hi);

  const __bf16* Kh = Kb + (size_t)h * TTOK * HPAD;
  const __bf16* Vh = Vt + (size_t)h * HPAD * TTOK;

  float mstate[8], lstate[8], alph[8];
  v8f o[6];
#pragma unroll
  for (int r = 0; r < 8; ++r) { mstate[r] = -1e30f; lstate[r] = 0.f; }
#pragma unroll
  for (int nc = 0; nc < 6; ++nc) o[nc] = (v8f){0.f,0.f,0.f,0.f,0.f,0.f,0.f,0.f};

  for (int kb = kstart; kb < kend; kb += 32) {
    v8f s0 = (v8f){0.f,0.f,0.f,0.f,0.f,0.f,0.f,0.f};
    v8f s1 = (v8f){0.f,0.f,0.f,0.f,0.f,0.f,0.f,0.f};
#pragma unroll
    for (int c = 0; c < 3; ++c) {
      v16bf bk0 = *(const v16bf*)(Kh + (size_t)(kb + nlo) * HPAD + c * 32 + hi * 16);
      v16bf bk1 = *(const v16bf*)(Kh + (size_t)(kb + 16 + nlo) * HPAD + c * 32 + hi * 16);
      s0 = wmma_bf16(aq[c], bk0, s0);
      s1 = wmma_bf16(aq[c], bk1, s1);
    }

    const int tk0 = kb + nlo;
    const int tk1 = kb + 16 + nlo;
    const bool in0 = (tk0 >= sfs) && (tk0 < kend);
    const bool in1 = (tk1 >= sfs) && (tk1 < kend);
    const bool hi0 = tk0 >= bnd;
    const bool hi1 = tk1 >= bnd;

#pragma unroll
    for (int r = 0; r < 8; ++r) {
      const float x0 = (in0 && (hi0 == rq[r])) ? s0[r] : -1e30f;
      const float x1 = (in1 && (hi1 == rq[r])) ? s1[r] : -1e30f;
      float mx = fmaxf(x0, x1);
#pragma unroll
      for (int off = 8; off > 0; off >>= 1) mx = fmaxf(mx, __shfl_xor(mx, off));
      const float mnew = fmaxf(mstate[r], mx);
      alph[r] = __expf(mstate[r] - mnew);
      const float p0 = __expf(x0 - mnew);
      const float p1 = __expf(x1 - mnew);
      s0[r] = p0;
      s1[r] = p1;
      float rs = p0 + p1;
#pragma unroll
      for (int off = 8; off > 0; off >>= 1) rs += __shfl_xor(rs, off);
      lstate[r] = lstate[r] * alph[r] + rs;
      mstate[r] = mnew;
    }

#pragma unroll
    for (int nc = 0; nc < 6; ++nc)
#pragma unroll
      for (int r = 0; r < 8; ++r) o[nc][r] *= alph[r];

    __syncthreads();
#pragma unroll
    for (int r = 0; r < 8; ++r) {
      sP[hi * 8 + r][nlo]      = (__bf16)s0[r];
      sP[hi * 8 + r][16 + nlo] = (__bf16)s1[r];
    }
    __syncthreads();

    union { v16bf v; float4 f[2]; } up;
    up.f[0] = *(const float4*)&sP[nlo][hi * 16];
    up.f[1] = *(const float4*)&sP[nlo][hi * 16 + 8];
    const v16bf ap = up.v;

#pragma unroll
    for (int nc = 0; nc < 6; ++nc) {
      v16bf bv = *(const v16bf*)(Vh + (size_t)(nc * 16 + nlo) * TTOK + kb + hi * 16);
      o[nc] = wmma_bf16(ap, bv, o[nc]);
    }
  }

  float rl[8];
#pragma unroll
  for (int r = 0; r < 8; ++r) rl[r] = 1.0f / lstate[r];
#pragma unroll
  for (int nc = 0; nc < 5; ++nc) {
    const int d = nc * 16 + nlo;
    if (d < HDIM) {
#pragma unroll
      for (int r = 0; r < 8; ++r) {
        const int t = q0 + hi * 8 + r;
        attnb[(size_t)t * EMBED + h * HDIM + d] = (__bf16)(o[nc][r] * rl[r]);
      }
    }
  }
}

extern "C" void kernel_launch(void* const* d_in, const int* in_sizes, int n_in,
                              void* d_out, int out_size, void* d_ws, size_t ws_size,
                              hipStream_t stream) {
  const float* x      = (const float*)d_in[0];   // [1,4096,1152]
  const float* w_qkv  = (const float*)d_in[1];   // [3456,1152]
  const float* b_qkv  = (const float*)d_in[2];   // [3456]
  const float* w_out  = (const float*)d_in[3];   // [1152,1152]
  const float* b_out  = (const float*)d_in[4];   // [1152]
  const int*   cu     = (const int*)d_in[5];     // [9]
  float* out = (float*)d_out;                    // [1,4096,1152]

  char* ws = (char*)d_ws;
  size_t off = 0;
  __bf16* xb    = (__bf16*)(ws + off); off += (size_t)TTOK * EMBED * 2;           // 9.44 MB
  __bf16* wqkvb = (__bf16*)(ws + off); off += (size_t)3 * EMBED * EMBED * 2;      // 7.96 MB
  __bf16* woutb = (__bf16*)(ws + off); off += (size_t)EMBED * EMBED * 2;          // 2.65 MB
  __bf16* qkvb  = (__bf16*)(ws + off); off += (size_t)TTOK * 3 * EMBED * 2;       // 28.3 MB
  __bf16* Qb    = (__bf16*)(ws + off); off += (size_t)NHEAD * TTOK * HPAD * 2;    // 12.6 MB
  __bf16* Kb    = (__bf16*)(ws + off); off += (size_t)NHEAD * TTOK * HPAD * 2;    // 12.6 MB
  __bf16* Vt    = (__bf16*)(ws + off); off += (size_t)NHEAD * HPAD * TTOK * 2;    // 12.6 MB
  __bf16* attnb = xb;  // x-bf16 is dead after GEMM1; reuse (same size)

  // 1) convert inputs to bf16
  f32_to_bf16_kernel<<<2048, 256, 0, stream>>>(x, xb, (size_t)TTOK * EMBED);
  f32_to_bf16_kernel<<<2048, 256, 0, stream>>>(w_qkv, wqkvb, (size_t)3 * EMBED * EMBED);
  f32_to_bf16_kernel<<<1024, 256, 0, stream>>>(w_out, woutb, (size_t)EMBED * EMBED);

  // 2) QKV projection: [4096,3456] = xb @ wqkv^T + b_qkv  (bf16 out)
  {
    dim3 grid(3 * EMBED / 128, TTOK / 128);
    gemm_lds_kernel<true><<<grid, 256, 0, stream>>>(xb, wqkvb, b_qkv, qkvb,
                                                    TTOK, 3 * EMBED, EMBED);
  }

  // 3) pack/pad/scale into attention layouts
  {
    const float scale = 0.11785113019775793f;  // 72^-0.5
    dim3 grid(TTOK, NHEAD);
    pack_qkv_kernel<<<grid, HPAD, 0, stream>>>(qkvb, Qb, Kb, Vt, scale);
  }

  // 4) block-diagonal flash attention
  {
    dim3 grid(TTOK / 16, NHEAD);
    attn_kernel<<<grid, 32, 0, stream>>>(Qb, Kb, Vt, cu, attnb);
  }

  // 5) output projection: [4096,1152] = attnb @ wout^T + b_out (fp32 out)
  {
    dim3 grid(EMBED / 128, TTOK / 128);
    gemm_lds_kernel<false><<<grid, 256, 0, stream>>>(attnb, woutb, b_out, out,
                                                     TTOK, EMBED, EMBED);
  }
}
